// Gemma4Attention_91199335563737
// MI455X (gfx1250) — matrix operand order repaired and checked by reading, not compiled, not run on hardware
//
#include <hip/hip_runtime.h>

#define B_SZ 2
#define T_SEQ 2048
#define D_MODEL 2048
#define NH 8
#define NKV 4
#define HD 256
#define SC 2048
#define WIN 512
#define SOFTCAP 50.0f
#define NEGINF (-2.3819763e+38f)

typedef __attribute__((ext_vector_type(16))) __bf16 v16bf;
typedef __attribute__((ext_vector_type(8)))  float  v8f;
typedef __attribute__((ext_vector_type(4))) unsigned int u32x4;
typedef __attribute__((ext_vector_type(8))) int i32x8;
typedef __attribute__((ext_vector_type(4))) int i32x4;

#if defined(__HIP_DEVICE_COMPILE__) && __has_builtin(__builtin_amdgcn_tensor_load_to_lds)
#define HAVE_TDM 1
#else
#define HAVE_TDM 0
#endif

// Scheduling-group directives: pin "all DS reads, then back-to-back WMMAs"
// pipelines (masks: 0x100 = DS_READ, 0x200 = DS_WRITE, 0x8 = MFMA/WMMA).
#if defined(__HIP_DEVICE_COMPILE__) && __has_builtin(__builtin_amdgcn_sched_group_barrier)
#define SGB(mask, n) __builtin_amdgcn_sched_group_barrier((mask), (n), 0)
#else
#define SGB(mask, n)
#endif

union U16 {
  unsigned short u[16];
  unsigned int   d[8];
  v16bf          v;
};

__device__ inline float bf2f(unsigned short u) {
  union { unsigned int i; float f; } v; v.i = ((unsigned int)u) << 16; return v.f;
}
__device__ inline unsigned short f2bf(float f) {
  union { float f; unsigned int i; } v; v.f = f;
  unsigned int i = v.i;
  unsigned int r = (i + 0x7fffu + ((i >> 16) & 1u)) >> 16;
  return (unsigned short)r;
}

__device__ inline v8f wmma_bf16(v16bf a, v16bf b, v8f c) {
  return __builtin_amdgcn_wmma_f32_16x16x32_bf16(false, a, false, b, (short)0, c,
                                                 false, false);
}

// A-fragment (16x32 bf16): lane holds row m=lane&15; lanes<16 get K={0..7,16..23},
// lanes>=16 get K={8..15,24..31}. `row` points at element [m][k_base].
__device__ inline v16bf load_a_frag(const unsigned short* row, int hi8) {
  U16 u;
  const unsigned int* p0 = (const unsigned int*)(row + hi8);
  const unsigned int* p1 = (const unsigned int*)(row + 16 + hi8);
  u.d[0] = p0[0]; u.d[1] = p0[1]; u.d[2] = p0[2]; u.d[3] = p0[3];
  u.d[4] = p1[0]; u.d[5] = p1[1]; u.d[6] = p1[2]; u.d[7] = p1[3];
  return u.v;
}

// B-fragment (32x16 bf16): lane holds column n=lane&15; elements = K in A-frag order
// (lanes<16: K={0..7,16..23}, lanes>=16: K={8..15,24..31}); `p` points at [k_base][n].
__device__ inline v16bf load_b_row(const unsigned short* p, int stride, int hi8) {
  U16 u;
  for (int s = 0; s < 8; ++s) {
    u.u[s] = p[(size_t)(hi8 + s) * stride];
    u.u[s + 8] = p[(size_t)(16 + hi8 + s) * stride];
  }
  return u.v;
}

__device__ inline unsigned lds_addr_of(const void* p) {
  // generic (flat) LDS pointer: addr[31:0] is the LDS byte offset
  return (unsigned)(unsigned long long)p;
}

#if HAVE_TDM
// TDM 2D tile load: tile_rows x row_elems (bf16) rows from a row-major tensor
// (row stride row_stride_elems) into LDS, padding pad_amount(+1) dwords every
// 2^(pad_interval+1) dwords. Rows >= tensor_rows read back zeros.
__device__ inline void tdm_load_2d_bf16(unsigned lds_byte_addr, const void* gsrc,
                                        unsigned tile_rows, unsigned row_elems,
                                        unsigned tensor_rows,
                                        unsigned row_stride_elems,
                                        unsigned pad_interval, unsigned pad_amount) {
  unsigned long long ga = (unsigned long long)gsrc;
  u32x4 g0;
  g0.x = 1u;                                                  // count=1 (valid)
  g0.y = lds_byte_addr;                                       // lds_addr
  g0.z = (unsigned)ga;                                        // global_addr lo
  g0.w = ((unsigned)(ga >> 32) & 0x01ffffffu) | (2u << 30);   // addr hi | type=2
  i32x8 g1;
  // data_size=1 (2B) | pad_enable | pad_interval | pad_amount
  g1[0] = (int)((1u << 16) | (1u << 20) | (pad_interval << 22) | (pad_amount << 25));
  g1[1] = (int)((row_elems & 0xffffu) << 16);                 // tensor_dim0 lo16
  g1[2] = (int)((row_elems >> 16) | ((tensor_rows & 0xffffu) << 16));
  g1[3] = (int)((tensor_rows >> 16) | ((row_elems & 0xffffu) << 16)); // tile_dim0
  g1[4] = (int)(tile_rows & 0xffffu);                         // tile_dim1
  g1[5] = (int)row_stride_elems;                              // dim0_stride lo32
  g1[6] = 0;
  g1[7] = 0;
  i32x4 z4 = {0, 0, 0, 0};
#if defined(__clang_major__) && (__clang_major__ >= 23)
  i32x8 z8 = {0, 0, 0, 0, 0, 0, 0, 0};
  __builtin_amdgcn_tensor_load_to_lds(g0, g1, z4, z4, z8, 0);
#else
  __builtin_amdgcn_tensor_load_to_lds(g0, g1, z4, z4, 0);
#endif
}
#endif

// ---------------- f32 -> bf16 convert ----------------
__global__ __launch_bounds__(256) void cvt_f32_bf16(const float* __restrict__ in,
                                                    unsigned short* __restrict__ out,
                                                    long long n) {
  long long i = (long long)blockIdx.x * blockDim.x + threadIdx.x;
  long long stride = (long long)gridDim.x * blockDim.x;
  for (; i < n; i += stride) out[i] = f2bf(in[i]);
}

// ---------------- bf16 WMMA GEMM: C[M,N] = A[M,K] * B[K,N] ----------------
// Block = 8 waves = 128x128 output tile. The shared 32x128 B tile is staged
// into LDS once per k-step by wave 0 via the Tensor Data Mover (double
// buffered, one barrier per iteration); A fragments are register
// double-buffered from global. Sched-group directives force: a group of 16 DS reads
// up front, then 8 back-to-back WMMAs.
#define BROW 136   // 128 elems + 8 pad (272B rows; TDM pad 4dw per 64dw)

__global__ __launch_bounds__(256) void gemm_bf16(const unsigned short* __restrict__ A,
                                                 const unsigned short* __restrict__ B,
                                                 float* __restrict__ Cf,
                                                 unsigned short* __restrict__ Cb,
                                                 int M, int N, int K, int store_bf) {
  __shared__ unsigned short lb[2][32 * BROW];
  const int tid = threadIdx.x;
  const int lane = tid & 31;
  const int wave = tid >> 5;
  const int ln = lane & 15;
  const int hi = lane >> 4;
  const int m0 = (blockIdx.x * 8 + wave) * 16;
  const int n0 = blockIdx.y * 128;

  v8f zero = {0.f, 0.f, 0.f, 0.f, 0.f, 0.f, 0.f, 0.f};
  v8f acc[8];
#pragma unroll
  for (int nc = 0; nc < 8; ++nc) acc[nc] = zero;

  const unsigned short* arow = A + (size_t)(m0 + ln) * K;

  auto stageB = [&](int buf, int k0) {
#if HAVE_TDM
    if (wave == 0)
      tdm_load_2d_bf16(lds_addr_of(&lb[buf][0]), B + (size_t)k0 * N + n0,
                       32, 128, (unsigned)(K - k0), (unsigned)N, 5, 3);
#else
    for (int idx = tid; idx < 32 * 16; idx += 256) {   // 512 x 16B chunks
      int kr = idx >> 4, ch = idx & 15;
      uint4 v = *(const uint4*)(B + (size_t)(k0 + kr) * N + n0 + ch * 8);
      *(uint4*)(&lb[buf][kr * BROW + ch * 8]) = v;
    }
#endif
  };

  stageB(0, 0);
  v16bf af = load_a_frag(arow, hi * 8);
#if HAVE_TDM
  __builtin_amdgcn_s_wait_tensorcnt(0);
#endif
  __syncthreads();

  for (int k0 = 0; k0 < K; k0 += 32) {
    const int buf = (k0 >> 5) & 1;
    const int k1 = k0 + 32;
    v16bf afn = af;
    if (k1 < K) {
      stageB(buf ^ 1, k1);                 // DMA overlaps the WMMAs below
      afn = load_a_frag(arow + k1, hi * 8);
    }
    const unsigned short* brow = &lb[buf][ln];
    v16bf bfr[8];
#pragma unroll
    for (int nc = 0; nc < 8; ++nc) bfr[nc] = load_b_row(brow + nc * 16, BROW, hi * 8);
#pragma unroll
    for (int nc = 0; nc < 8; ++nc) acc[nc] = wmma_bf16(af, bfr[nc], acc[nc]);
    SGB(0x100, 16);                        // all 16 DS reads first
    SGB(0x008, 8);                         // then 8 WMMAs back-to-back
    af = afn;
    if (k1 < K) {
#if HAVE_TDM
      __builtin_amdgcn_s_wait_tensorcnt(0); // wave0: next tile landed
#endif
      __syncthreads();
    }
  }

#pragma unroll
  for (int nc = 0; nc < 8; ++nc) {
#pragma unroll
    for (int r = 0; r < 8; ++r) {
      size_t m = (size_t)(m0 + r + 8 * hi);
      size_t n = (size_t)(n0 + nc * 16 + ln);
      if (store_bf) Cb[m * N + n] = f2bf(acc[nc][r]);
      else          Cf[m * N + n] = acc[nc][r];
    }
  }
}

// ---------------- in-place RMSNorm + RoPE on a bf16 (…, heads, 256) buffer ----
__global__ __launch_bounds__(128) void normrope(unsigned short* __restrict__ qk,
                                                const float* __restrict__ nscale,
                                                const int* __restrict__ segpos,
                                                int n_heads) {
  const int vid = blockIdx.x;                      // b*T*nh + t*nh + h
  const int t = (vid / n_heads) % T_SEQ;
  const int b = vid / (n_heads * T_SEQ);
  unsigned short* p = qk + (size_t)vid * HD;
  const int i = threadIdx.x;                       // 0..127

  float x1 = bf2f(p[i]);
  float x2 = bf2f(p[i + 128]);

  __shared__ float red[128];
  red[i] = x1 * x1 + x2 * x2;
  __syncthreads();
  for (int s = 64; s > 0; s >>= 1) {
    if (i < s) red[i] += red[i + s];
    __syncthreads();
  }
  float r = rsqrtf(red[0] * (1.0f / HD) + 1e-6f);
  float n1 = x1 * r * (1.0f + nscale[i]);
  float n2 = x2 * r * (1.0f + nscale[i + 128]);

  float pos = (float)segpos[b * T_SEQ + t];
  // 1/timescale = 10000^(-i/128) = exp2(-i * log2(10000)/128)
  float invts = exp2f((float)i * (-13.287712379549449f / 128.0f));
  float ang = pos * invts;
  float sa = sinf(ang), ca = cosf(ang);
  p[i]       = f2bf(n1 * ca - n2 * sa);
  p[i + 128] = f2bf(n2 * ca + n1 * sa);
}

// ---------------- flash attention: one wave per (b, head, 16 t-rows) ----------
// K/V 32x256 tiles staged to LDS via TDM (double buffered, TENSORcnt-tracked),
// with TDM LDS padding for conflict-free rows. Sched-group directives pin
// 8-DS-read / 4-WMMA pipelines in the QK and PV phases.
#define PSTRIDE 34
#define KV_ROWS 32
#define LROW 264   // 256 elems + 8 pad (528B row stride; TDM pad 4dw/128dw)

__global__ __launch_bounds__(32) void attn_fa(const unsigned short* __restrict__ Qb,
                                              const unsigned short* __restrict__ Kb,
                                              const unsigned short* __restrict__ Vb,
                                              unsigned short* __restrict__ Ob) {
  __shared__ unsigned short lk[2][KV_ROWS * LROW];
  __shared__ unsigned short lv[2][KV_ROWS * LROW];
  __shared__ unsigned short pl[16 * PSTRIDE];      // P tile, [t][s]

  const int lane = threadIdx.x & 31;
  const int ln = lane & 15;
  const int hi = lane >> 4;
  const int t0 = blockIdx.x * 16;
  const int h  = blockIdx.y;
  const int kv = h / (NH / NKV);
  const int b  = blockIdx.z;

  const size_t qstride = (size_t)NH * HD;
  const size_t kstride = (size_t)NKV * HD;
  const unsigned short* Q = Qb + ((size_t)b * T_SEQ) * qstride + (size_t)h * HD;
  const unsigned short* K = Kb + ((size_t)b * SC) * kstride + (size_t)kv * HD;
  const unsigned short* V = Vb + ((size_t)b * SC) * kstride + (size_t)kv * HD;

  // Preload Q as B-fragments (lane = t row t0+ln, elements = head-dim K in A-frag order).
  v16bf qfrag[8];
#pragma unroll
  for (int c = 0; c < 8; ++c) {
    U16 u;
    const unsigned short* qp = Q + (size_t)(t0 + ln) * qstride + c * 32;
    u.v = load_a_frag(qp, hi * 8);
    qfrag[c] = u.v;
  }

  v8f zero = {0.f, 0.f, 0.f, 0.f, 0.f, 0.f, 0.f, 0.f};
  v8f O[16];
#pragma unroll
  for (int c = 0; c < 16; ++c) O[c] = zero;
  float rowmax = NEGINF;   // stats for row t = t0 + ln (duplicated across hi)
  float rowsum = 0.0f;

  int s_begin = t0 - (WIN - 1);
  if (s_begin < 0) s_begin = 0;
  s_begin &= ~31;
  const int s_end = t0 + 15;
  const int nsteps = (s_end - s_begin) / 32 + 1;
  const float qscale = 0.0625f;                    // HD^-0.5

  auto stage = [&](int buf, int s0) {
#if HAVE_TDM
    unsigned tr = (unsigned)(SC - s0);             // rows left in tensor
    tdm_load_2d_bf16(lds_addr_of(&lk[buf][0]), K + (size_t)s0 * kstride,
                     KV_ROWS, HD, tr, (unsigned)kstride, 6, 3);
    tdm_load_2d_bf16(lds_addr_of(&lv[buf][0]), V + (size_t)s0 * kstride,
                     KV_ROWS, HD, tr, (unsigned)kstride, 6, 3);
#else
    for (int i = lane; i < KV_ROWS * (HD / 8); i += 32) {  // 16B chunks
      int row = i / (HD / 8), ch = i % (HD / 8);
      int s = s0 + row;
      uint4 kd = {0, 0, 0, 0}, vd = {0, 0, 0, 0};
      if (s < SC) {
        kd = *(const uint4*)(K + (size_t)s * kstride + ch * 8);
        vd = *(const uint4*)(V + (size_t)s * kstride + ch * 8);
      }
      *(uint4*)(&lk[buf][row * LROW + ch * 8]) = kd;
      *(uint4*)(&lv[buf][row * LROW + ch * 8]) = vd;
    }
#endif
  };

  stage(0, s_begin);

  for (int it = 0; it < nsteps; ++it) {
    const int s0 = s_begin + it * 32;
    const int buf = it & 1;
    if (it + 1 < nsteps) {
      stage(buf ^ 1, s0 + 32);
#if HAVE_TDM
      __builtin_amdgcn_s_wait_tensorcnt(2);        // current tiles complete
#endif
    } else {
#if HAVE_TDM
      __builtin_amdgcn_s_wait_tensorcnt(0);
#endif
    }
#if !HAVE_TDM
    __syncthreads();
#endif

    // S^T tiles: A = K rows (from LDS), B = preloaded Q fragments.
    v8f L0 = zero, L1 = zero;
    const unsigned short* krow0 = &lk[buf][ln * LROW];
    const unsigned short* krow1 = &lk[buf][(ln + 16) * LROW];
#pragma unroll
    for (int g = 0; g < 2; ++g) {
      v16bf ka[4];
#pragma unroll
      for (int j = 0; j < 4; ++j) ka[j] = load_a_frag(krow0 + (4 * g + j) * 32, hi * 8);
#pragma unroll
      for (int j = 0; j < 4; ++j) L0 = wmma_bf16(ka[j], qfrag[4 * g + j], L0);
#pragma unroll
      for (int j = 0; j < 4; ++j) ka[j] = load_a_frag(krow1 + (4 * g + j) * 32, hi * 8);
#pragma unroll
      for (int j = 0; j < 4; ++j) L1 = wmma_bf16(ka[j], qfrag[4 * g + j], L1);
    }
    SGB(0x100, 8); SGB(0x008, 4);   // QK pipeline: 8 DS reads, 4 WMMAs (x4)
    SGB(0x100, 8); SGB(0x008, 4);
    SGB(0x100, 8); SGB(0x008, 4);
    SGB(0x100, 8); SGB(0x008, 4);

    // softcap + sliding-window mask; element r: s = s0 + r + 8*hi (+16), t = t0+ln
    const int t = t0 + ln;
    float l0[8], l1[8];
    float mstep = NEGINF;
#pragma unroll
    for (int r = 0; r < 8; ++r) {
      int sa = s0 + r + 8 * hi;
      int sb = sa + 16;
      float a0 = SOFTCAP * tanhf(L0[r] * qscale * (1.0f / SOFTCAP));
      float a1 = SOFTCAP * tanhf(L1[r] * qscale * (1.0f / SOFTCAP));
      bool v0 = (sa <= t) && (t - sa < WIN);
      bool v1 = (sb <= t) && (t - sb < WIN);
      l0[r] = v0 ? a0 : NEGINF;
      l1[r] = v1 ? a1 : NEGINF;
      mstep = fmaxf(mstep, fmaxf(l0[r], l1[r]));
    }
    mstep = fmaxf(mstep, __shfl_xor(mstep, 16, 32));
    float mnew = fmaxf(rowmax, mstep);
    float mq   = fmaxf(mnew, -1e30f);
    float corr = expf(fmaxf(rowmax, -1e30f) - mq);
    float psum = 0.0f;
#pragma unroll
    for (int r = 0; r < 8; ++r) {
      float p0 = expf(l0[r] - mq);
      float p1 = expf(l1[r] - mq);
      psum += p0 + p1;
      pl[ln * PSTRIDE + (r + 8 * hi)]      = f2bf(p0);
      pl[ln * PSTRIDE + 16 + (r + 8 * hi)] = f2bf(p1);
    }
    psum += __shfl_xor(psum, 16, 32);
    rowsum = rowsum * corr + psum;
    rowmax = mnew;
    __syncthreads();

    // rescale O and accumulate P*V (V fragments from LDS, batched by 4)
    float corr_m[8];
#pragma unroll
    for (int r = 0; r < 8; ++r) corr_m[r] = __shfl(corr, r + 8 * hi, 32);

    v16bf pa = load_a_frag((const unsigned short*)pl + ln * PSTRIDE, hi * 8);
    const unsigned short* vrow = &lv[buf][ln];
#pragma unroll
    for (int g = 0; g < 4; ++g) {
      v16bf vf[4];
#pragma unroll
      for (int j = 0; j < 4; ++j) vf[j] = load_b_row(vrow + (4 * g + j) * 16, LROW, hi * 8);
#pragma unroll
      for (int j = 0; j < 4; ++j) {
        const int c = 4 * g + j;
#pragma unroll
        for (int r = 0; r < 8; ++r) O[c][r] *= corr_m[r];
        O[c] = wmma_bf16(pa, vf[j], O[c]);
      }
    }
    SGB(0x100, 8); SGB(0x008, 4);   // PV pipeline: 8 DS reads, 4 WMMAs (x4)
    SGB(0x100, 8); SGB(0x008, 4);
    SGB(0x100, 8); SGB(0x008, 4);
    SGB(0x100, 8); SGB(0x008, 4);
    __syncthreads();
  }

  // normalize and store bf16 (b, t, h, hd)
  float inv[8];
#pragma unroll
  for (int r = 0; r < 8; ++r) {
    float s = __shfl(rowsum, r + 8 * hi, 32);
    inv[r] = (s > 0.0f) ? (1.0f / s) : 0.0f;
  }
#pragma unroll
  for (int c = 0; c < 16; ++c) {
#pragma unroll
    for (int r = 0; r < 8; ++r) {
      int t = t0 + r + 8 * hi;
      Ob[(((size_t)b * T_SEQ + t) * NH + h) * HD + c * 16 + ln] = f2bf(O[c][r] * inv[r]);
    }
  }
}

// ---------------- host-side orchestration ----------------
extern "C" void kernel_launch(void* const* d_in, const int* in_sizes, int n_in,
                              void* d_out, int out_size, void* d_ws, size_t ws_size,
                              hipStream_t stream) {
  const float* x      = (const float*)d_in[0];
  const int*   segpos = (const int*)d_in[1];
  // d_in[2] = cur_ind (scalar 0; cache fully overwritten since T == S_CACHE)
  const float* wq  = (const float*)d_in[3];
  const float* wk  = (const float*)d_in[4];
  const float* wv  = (const float*)d_in[5];
  const float* wo  = (const float*)d_in[6];
  const float* qns = (const float*)d_in[7];
  const float* kns = (const float*)d_in[8];
  float* out = (float*)d_out;

  char* ws = (char*)d_ws;
  size_t off = 0;
  auto alloc = [&](size_t elems) -> unsigned short* {
    unsigned short* p = (unsigned short*)(ws + off);
    off += ((elems * 2 + 255) & ~(size_t)255);
    return p;
  };
  const size_t MT = (size_t)B_SZ * T_SEQ;                 // 4096 rows

  unsigned short* xb  = alloc(MT * D_MODEL);
  unsigned short* wqb = alloc((size_t)D_MODEL * NH * HD);
  unsigned short* wkb = alloc((size_t)D_MODEL * NKV * HD);
  unsigned short* wvb = alloc((size_t)D_MODEL * NKV * HD);
  unsigned short* wob = alloc((size_t)NH * HD * D_MODEL);
  unsigned short* qb  = alloc(MT * NH * HD);
  unsigned short* kb  = alloc(MT * NKV * HD);
  unsigned short* vb  = alloc(MT * NKV * HD);
  unsigned short* ab  = alloc(MT * NH * HD);              // total ~92 MB

  // 1) convert activations + weights to bf16
  cvt_f32_bf16<<<2048, 256, 0, stream>>>(x,  xb,  (long long)(MT * D_MODEL));
  cvt_f32_bf16<<<2048, 256, 0, stream>>>(wq, wqb, (long long)D_MODEL * NH * HD);
  cvt_f32_bf16<<<1024, 256, 0, stream>>>(wk, wkb, (long long)D_MODEL * NKV * HD);
  cvt_f32_bf16<<<1024, 256, 0, stream>>>(wv, wvb, (long long)D_MODEL * NKV * HD);
  cvt_f32_bf16<<<2048, 256, 0, stream>>>(wo, wob, (long long)NH * HD * D_MODEL);

  // 2) QKV projections (WMMA + TDM-staged B tiles), bf16 outputs
  gemm_bf16<<<dim3(32, (NH * HD) / 128), 256, 0, stream>>>(
      xb, wqb, nullptr, qb, (int)MT, NH * HD, D_MODEL, 1);
  gemm_bf16<<<dim3(32, (NKV * HD) / 128), 256, 0, stream>>>(
      xb, wkb, nullptr, kb, (int)MT, NKV * HD, D_MODEL, 1);
  gemm_bf16<<<dim3(32, (NKV * HD) / 128), 256, 0, stream>>>(
      xb, wvb, nullptr, vb, (int)MT, NKV * HD, D_MODEL, 1);

  // 3) RMSNorm + RoPE in place on q and k
  normrope<<<B_SZ * T_SEQ * NH,  128, 0, stream>>>(qb, qns, segpos, NH);
  normrope<<<B_SZ * T_SEQ * NKV, 128, 0, stream>>>(kb, kns, segpos, NKV);

  // 4) sliding-window flash attention (WMMA + TDM double buffering)
  attn_fa<<<dim3(T_SEQ / 16, NH, B_SZ), 32, 0, stream>>>(qb, kb, vb, ab);

  // 5) output projection (WMMA + TDM-staged B tiles), fp32 into d_out
  gemm_bf16<<<dim3(32, D_MODEL / 128), 256, 0, stream>>>(
      ab, wob, out, nullptr, (int)MT, D_MODEL, NH * HD, 0);
}
